// Mamba2Mixer_69123203662684
// MI455X (gfx1250) — matrix operand order repaired and checked by reading, not compiled, not run on hardware
//
#include <hip/hip_runtime.h>
#include <hip/hip_bf16.h>
#include <stdint.h>

// ---------------- problem constants ----------------
#define SEQL    4096
#define DMODEL  1024
#define DINNER  2048
#define NHEADS  32
#define HEADDIM 64
#define DSTATE  128
#define CONVDIM 2304          // DINNER + 2*DSTATE
#define DINPROJ 4384          // 2*DINNER + 2*DSTATE + NHEADS
#define NPAD    4480          // DINPROJ padded up to a multiple of 128
#define LNEPS   1e-5f

typedef __bf16 bf16;
typedef __attribute__((ext_vector_type(16))) __bf16 v16bf;
typedef __attribute__((ext_vector_type(8)))  __bf16 v8bf;
typedef __attribute__((ext_vector_type(8)))  float  v8f;
typedef int v4i __attribute__((vector_size(4 * sizeof(int))));

// ---------------- CDNA5 async global->LDS (probe-confirmed present) ----------------
#if defined(__AMDGCN__) && __has_builtin(__builtin_amdgcn_global_load_async_to_lds_b128)
#define USE_ASYNC_LDS 1
#else
#define USE_ASYNC_LDS 0
#endif

__device__ __forceinline__ void copy_b128_to_lds(const void* g, void* l) {
#if USE_ASYNC_LDS
  __builtin_amdgcn_global_load_async_to_lds_b128(
      (__attribute__((address_space(1))) v4i*)g,
      (__attribute__((address_space(3))) v4i*)l, 0, 0);
#else
  *(uint4*)l = *(const uint4*)g;
#endif
}

__device__ __forceinline__ void wait_lds_copies() {
#if USE_ASYNC_LDS
#if __has_builtin(__builtin_amdgcn_s_wait_asynccnt)
  __builtin_amdgcn_s_wait_asynccnt(0);
#else
  asm volatile("s_wait_asynccnt 0" ::: "memory");
#endif
#endif
}

// ---------------- generic NT bf16 WMMA GEMM ----------------
// C[M,N](f32) = A[M,K](bf16, row-major) * B[N,K](bf16, row-major)^T
// M%128, N%128, K%32 == 0 by construction (padding) -> no bounds checks.
// blockDim = 256 (8 wave32s), block tile 128x128, BK=32.
// Wave grid 4(m) x 2(n); each wave computes 32x64 via 2x4 WMMA frags.
// Double-buffered LDS: async loads for tile i+1 are issued before the WMMAs
// of tile i, hiding HBM latency; one barrier per K-step.
// SPLITK (template) accumulates partials with f32 atomics over blockIdx.z.
#define BM 128
#define BN 128
#define BK 32
#define BKP 40   // padded LDS row (elements) -> 80B rows, 16B-aligned fragments

template <bool SPLITK>
__global__ __launch_bounds__(256) void gemm_nt_bf16_kernel(
    const bf16* __restrict__ A, const bf16* __restrict__ B, float* __restrict__ C,
    int M, int N, int K)
{
  (void)M;
  __shared__ bf16 As[2][BM * BKP];
  __shared__ bf16 Bs[2][BN * BKP];

  const int tid  = threadIdx.x;
  const int lane = tid & 31;
  const int wave = tid >> 5;
  const int wm   = wave >> 1;     // 0..3  -> 32-row slab
  const int wn   = wave & 1;      // 0..1  -> 64-col slab
  const long tileM = (long)blockIdx.y * BM;
  const long tileN = (long)blockIdx.x * BN;

  const int r  = lane & 15;       // row/col within fragment
  const int hs = lane >> 4;       // half-select per ISA A/B layouts

  const int kChunk = K / (int)gridDim.z;
  const int kBeg   = kChunk * (int)blockIdx.z;
  const int nIter  = kChunk / BK;

  v8f acc[2][4];
#pragma unroll
  for (int i = 0; i < 2; ++i)
#pragma unroll
    for (int j = 0; j < 4; ++j) { v8f z = {0.f,0.f,0.f,0.f,0.f,0.f,0.f,0.f}; acc[i][j] = z; }

  // global->LDS staging: 256 threads x 32B per operand tile (128x32 bf16)
  const int crow  = tid >> 1;          // 0..127 tile row
  const int celem = (tid & 1) * 16;    // element offset 0 / 16 within 32-elem row

  const bf16* gArow = A + (size_t)(tileM + crow) * K + celem;
  const bf16* gBrow = B + (size_t)(tileN + crow) * K + celem;
  const int   lOff  = crow * BKP + celem;

  auto issue_tile = [&](int buf, int k0) {
    copy_b128_to_lds(gArow + k0,     &As[buf][lOff]);
    copy_b128_to_lds(gArow + k0 + 8, &As[buf][lOff + 8]);
    copy_b128_to_lds(gBrow + k0,     &Bs[buf][lOff]);
    copy_b128_to_lds(gBrow + k0 + 8, &Bs[buf][lOff + 8]);
  };

  int buf = 0;
  issue_tile(0, kBeg);                 // prologue prefetch

  for (int it = 0; it < nIter; ++it) {
    wait_lds_copies();                 // tile `it` landed (this wave's asyncs)
    __syncthreads();                   // all waves' tiles landed / prev reads done
    if (it + 1 < nIter)
      issue_tile(buf ^ 1, kBeg + (it + 1) * BK);  // prefetch next tile behind WMMAs

    union AB { v16bf v; v8bf h[2]; };
    AB afrag[2];
#pragma unroll
    for (int fm = 0; fm < 2; ++fm) {
      // 16-bit A 16x32 layout: lanes<16 K=0..7 & 16..23; lanes>=16 K=8..15 & 24..31
      const bf16* ap = &As[buf][(wm * 32 + fm * 16 + r) * BKP];
      afrag[fm].h[0] = *(const v8bf*)(ap + hs * 8);
      afrag[fm].h[1] = *(const v8bf*)(ap + 16 + hs * 8);
    }
#pragma unroll
    for (int fn = 0; fn < 4; ++fn) {
      // 16-bit B 32x16 layout: lanes<16 K=0..7 & 16..23 of col (lane&15); lanes>=16 K=8..15 & 24..31
      AB bfrag;
      const bf16* bp = &Bs[buf][(wn * 64 + fn * 16 + r) * BKP + hs * 8];
      bfrag.h[0] = *(const v8bf*)(bp);
      bfrag.h[1] = *(const v8bf*)(bp + 16);
#pragma unroll
      for (int fm = 0; fm < 2; ++fm) {
        acc[fm][fn] = __builtin_amdgcn_wmma_f32_16x16x32_bf16(
            false, afrag[fm].v, false, bfrag.v, (short)0, acc[fm][fn], false, false);
      }
    }
    buf ^= 1;
  }

  // epilogue: C/D layout -> VGPR e, lane: M = e + 8*hs, N = lane&15 (branch-free)
#pragma unroll
  for (int fm = 0; fm < 2; ++fm) {
    const long row0 = tileM + wm * 32 + fm * 16 + hs * 8;
#pragma unroll
    for (int fn = 0; fn < 4; ++fn) {
      const long col = tileN + wn * 64 + fn * 16 + r;
      float* cp = C + row0 * (long)N + col;
#pragma unroll
      for (int e = 0; e < 8; ++e) {
        if (SPLITK) atomicAdd(cp + (size_t)e * N, acc[fm][fn][e]);
        else        cp[(size_t)e * N] = acc[fm][fn][e];
      }
    }
  }
}

// ---------------- elementwise kernels ----------------
__global__ void cast_pad_kernel(const float* __restrict__ src, bf16* __restrict__ dst,
                                long n_src, long n_total) {
  long i = (long)blockIdx.x * blockDim.x + threadIdx.x;
  if (i < n_total) dst[i] = (i < n_src) ? (bf16)src[i] : (bf16)0.f;
}

__global__ void fill_zero_kernel(float* __restrict__ p, long n) {
  long i = (long)blockIdx.x * blockDim.x + threadIdx.x;
  if (i < n) p[i] = 0.f;
}

// dA[l,h] = softplus(zx[l, -32+h] + dt_bias[h]) * (-exp(A_log[h]))
__global__ void dt_da_kernel(const float* __restrict__ zx, const float* __restrict__ dt_bias,
                             const float* __restrict__ A_log, float* __restrict__ dA) {
  int i = blockIdx.x * blockDim.x + threadIdx.x;
  if (i >= SEQL * NHEADS) return;
  int l = i / NHEADS, h = i % NHEADS;
  float v  = zx[(size_t)l * NPAD + (DINPROJ - NHEADS) + h] + dt_bias[h];
  float sp = fmaxf(v, 0.f) + log1pf(expf(-fabsf(v)));    // stable softplus
  dA[i] = sp * (-expf(A_log[h]));
}

// depthwise 4-tap conv (SAME: pad_lo=1) + bias + SiLU, routed into GEMM layouts:
//   c <  2048          : xin(f32 [L,2048]) and Vs^T(bf16 [2048,L], scaled by dA)
//   2048 <= c < 2176   : B^T (bf16 [128,L])
//   c >= 2176          : C   (bf16 [L,128])
__global__ void conv_silu_kernel(const float* __restrict__ zx, const float* __restrict__ cw,
                                 const float* __restrict__ cb, const float* __restrict__ dA,
                                 float* __restrict__ xin, bf16* __restrict__ vst,
                                 bf16* __restrict__ bt, bf16* __restrict__ cm) {
  long i = (long)blockIdx.x * blockDim.x + threadIdx.x;
  if (i >= (long)SEQL * CONVDIM) return;
  int c = (int)(i % CONVDIM);
  int l = (int)(i / CONVDIM);
  float acc = cb[c];
  const float* col = zx + DINNER + c;
#pragma unroll
  for (int j = 0; j < 4; ++j) {
    int li = l + j - 1;
    if (li >= 0 && li < SEQL) acc += cw[c * 4 + j] * col[(size_t)li * NPAD];
  }
  float s = acc / (1.f + expf(-acc));   // SiLU
  if (c < DINNER) {
    int h = c >> 6;
    xin[(size_t)l * DINNER + c] = s;
    vst[(size_t)c * SEQL + l]   = (bf16)(s * dA[l * NHEADS + h]);
  } else if (c < DINNER + DSTATE) {
    bt[(size_t)(c - DINNER) * SEQL + l] = (bf16)s;
  } else {
    cm[(size_t)l * DSTATE + (c - DINNER - DSTATE)] = (bf16)s;
  }
}

// per-row LayerNorm over 2048 (+ D*x_in residual) then SiLU(z) gate -> bf16
__global__ __launch_bounds__(256) void ln_gate_kernel(
    const float* __restrict__ yraw, const float* __restrict__ xin,
    const float* __restrict__ Dv, const float* __restrict__ lnw,
    const float* __restrict__ lnb, const float* __restrict__ zx,
    bf16* __restrict__ yg) {
  const int l = blockIdx.x;
  const int tid = threadIdx.x;
  float yv[8];
  float s1 = 0.f, s2 = 0.f;
#pragma unroll
  for (int j = 0; j < 8; ++j) {
    int c = tid + j * 256;
    float y = yraw[(size_t)l * DINNER + c] + xin[(size_t)l * DINNER + c] * Dv[c >> 6];
    yv[j] = y; s1 += y; s2 += y * y;
  }
  __shared__ float r1[256], r2[256];
  r1[tid] = s1; r2[tid] = s2;
  __syncthreads();
  for (int off = 128; off > 0; off >>= 1) {
    if (tid < off) { r1[tid] += r1[tid + off]; r2[tid] += r2[tid + off]; }
    __syncthreads();
  }
  float mu   = r1[0] * (1.f / DINNER);
  float var  = r2[0] * (1.f / DINNER) - mu * mu;
  float rstd = rsqrtf(var + LNEPS);
#pragma unroll
  for (int j = 0; j < 8; ++j) {
    int c  = tid + j * 256;
    float yn = (yv[j] - mu) * rstd * lnw[c] + lnb[c];
    float z  = zx[(size_t)l * NPAD + c];
    float g  = z / (1.f + expf(-z));
    yg[(size_t)l * DINNER + c] = (bf16)(yn * g);
  }
}

// ---------------- host orchestration ----------------
extern "C" void kernel_launch(void* const* d_in, const int* in_sizes, int n_in,
                              void* d_out, int out_size, void* d_ws, size_t ws_size,
                              hipStream_t stream) {
  (void)in_sizes; (void)n_in; (void)out_size; (void)ws_size;
  const float* x       = (const float*)d_in[0];
  const float* W_in    = (const float*)d_in[1];
  const float* conv_w  = (const float*)d_in[2];
  const float* conv_b  = (const float*)d_in[3];
  const float* dt_bias = (const float*)d_in[4];
  const float* A_log   = (const float*)d_in[5];
  const float* Dv      = (const float*)d_in[6];
  const float* ln_w    = (const float*)d_in[7];
  const float* ln_b    = (const float*)d_in[8];
  const float* W_out   = (const float*)d_in[9];
  float* out = (float*)d_out;

  char* ws = (char*)d_ws;
  size_t off = 0;
  auto alloc = [&](size_t bytes) -> char* {
    char* p = ws + off;
    off = (off + bytes + 255) & ~(size_t)255;
    return p;
  };

  bf16*  xb    = (bf16*) alloc((size_t)SEQL   * DMODEL * 2);  // x in bf16
  bf16*  winb  = (bf16*) alloc((size_t)NPAD   * DMODEL * 2);  // W_in bf16, N-padded w/ zeros
  bf16*  woutb = (bf16*) alloc((size_t)DMODEL * DINNER * 2);  // W_out bf16
  float* zx    = (float*)alloc((size_t)SEQL   * NPAD   * 4);  // zxbcdt (ld = NPAD)
  float* dA    = (float*)alloc((size_t)SEQL   * NHEADS * 4);
  float* xin   = (float*)alloc((size_t)SEQL   * DINNER * 4);  // silu(conv) x-part
  bf16*  vst   = (bf16*) alloc((size_t)DINNER * SEQL   * 2);  // Vs^T
  bf16*  btm   = (bf16*) alloc((size_t)DSTATE * SEQL   * 2);  // B^T
  bf16*  cm    = (bf16*) alloc((size_t)SEQL   * DSTATE * 2);  // C
  float* kvt   = (float*)alloc((size_t)DINNER * DSTATE * 4);  // KV^T (f32, split-K acc)
  bf16*  kvtb  = (bf16*) alloc((size_t)DINNER * DSTATE * 2);
  float* yraw  = (float*)alloc((size_t)SEQL   * DINNER * 4);
  bf16*  yg    = (bf16*) alloc((size_t)SEQL   * DINNER * 2);  // gated, normalized y

  const int T = 256;
  auto nb = [](long n) { return (unsigned)((n + 255) / 256); };

  // casts (pad W_in rows 4384..4479 with zeros -> no bounds checks in GEMM1)
  cast_pad_kernel<<<nb((long)SEQL * DMODEL), T, 0, stream>>>(x, xb, (long)SEQL * DMODEL, (long)SEQL * DMODEL);
  cast_pad_kernel<<<nb((long)NPAD * DMODEL), T, 0, stream>>>(W_in, winb, (long)DINPROJ * DMODEL, (long)NPAD * DMODEL);
  cast_pad_kernel<<<nb((long)DMODEL * DINNER), T, 0, stream>>>(W_out, woutb, (long)DMODEL * DINNER, (long)DMODEL * DINNER);

  // G1: zxbcdt = X * W_in^T     [4096 x 4480] (padded), K=1024
  dim3 g1(NPAD / BN, SEQL / BM, 1);
  gemm_nt_bf16_kernel<false><<<g1, T, 0, stream>>>(xb, winb, zx, SEQL, NPAD, DMODEL);

  // dt -> dA, then depthwise conv + SiLU with routed/transposed outputs
  dt_da_kernel<<<nb((long)SEQL * NHEADS), T, 0, stream>>>(zx, dt_bias, A_log, dA);
  conv_silu_kernel<<<nb((long)SEQL * CONVDIM), T, 0, stream>>>(zx, conv_w, conv_b, dA, xin, vst, btm, cm);

  // G2: KV^T = Vs^T * (B^T)^T   [2048 x 128], K=4096, split-K=8 with f32 atomics
  fill_zero_kernel<<<nb((long)DINNER * DSTATE), T, 0, stream>>>(kvt, (long)DINNER * DSTATE);
  dim3 g2(DSTATE / BN, DINNER / BM, 8);
  gemm_nt_bf16_kernel<true><<<g2, T, 0, stream>>>(vst, btm, kvt, DINNER, DSTATE, SEQL);
  cast_pad_kernel<<<nb((long)DINNER * DSTATE), T, 0, stream>>>(kvt, kvtb, (long)DINNER * DSTATE, (long)DINNER * DSTATE);

  // G3: yraw = C * (KV^T)^T     [4096 x 2048], K=128
  dim3 g3(DINNER / BN, SEQL / BM, 1);
  gemm_nt_bf16_kernel<false><<<g3, T, 0, stream>>>(cm, kvtb, yraw, SEQL, DINNER, DSTATE);

  // LayerNorm(+D*x_in) * SiLU(z) gate -> bf16
  ln_gate_kernel<<<SEQL, T, 0, stream>>>(yraw, xin, Dv, ln_w, ln_b, zx, yg);

  // G4: out = yg * W_out^T      [4096 x 1024], K=2048  (writes d_out, f32)
  dim3 g4(DMODEL / BN, SEQL / BM, 1);
  gemm_nt_bf16_kernel<false><<<g4, T, 0, stream>>>(yg, woutb, out, SEQL, DMODEL, DINNER);
}
